// BiLSTM_CRF_76209899700295
// MI455X (gfx1250) — matrix operand order repaired and checked by reading, not compiled, not run on hardware
//
#include <hip/hip_runtime.h>
#include <hip/hip_bf16.h>

// ---------------------------------------------------------------------------
// BiLSTM-CRF for MI455X (gfx1250, wave32).
//
// Roofline: ~94 GFLOP total (input GEMMs 47 GF fully parallel, recurrence
// 47 GF latency-bound across 8 persistent WGs), ~120 MB unique data.
// Strategy: bf16 WMMA (v_wmma_f32_16x16x32_bf16) with f32 accumulation for
// all matrix work; f32 VALU for the tiny K=9 CRF/Viterbi dynamic programs.
//
// d_ws layout (bytes, 256-aligned), total ~420 MB:
//   WihP  : 2*1200*320*2   packed bf16 input weights   (p = 4*j + gate)
//   WhhP  : 2*1200*320*2   packed bf16 recurrent weights
//   BiasP : 2*1200*4       packed f32 (b_ih + b_hh)
//   Xbf   : 32768*320*2    bf16 embedded inputs, K padded 300->320
//   Gates : 2*32768*1200*4 f32  x@W_ih^T (packed cols), physical time
//   Hout  : 2*32768*300*4  f32  per-direction h, physical time
//   Emis  : 32768*9*4      f32  emissions
//   Hist  : 511*64*9*4     i32  viterbi backpointers
// ---------------------------------------------------------------------------

typedef __attribute__((ext_vector_type(8)))  __bf16 v8bf;
typedef __attribute__((ext_vector_type(16))) __bf16 v16bf;
typedef __attribute__((ext_vector_type(8)))  float  v8f;

namespace {
constexpr int V_  = 32000;
constexpr int D_  = 300;
constexpr int H_  = 300;
constexpr int K_  = 9;
constexpr int B_  = 64;
constexpr int T_  = 512;
constexpr int G4  = 4 * H_;      // 1200 packed gate rows
constexpr int DP  = 320;         // padded reduction dim (mult of 32)
constexpr int MR  = B_ * T_;     // 32768 rows
constexpr int NT  = G4 / 16;     // 75 N-tiles
constexpr int KT  = DP / 32;     // 10 K-steps per WMMA chain
}

__device__ __forceinline__ float sigf(float x) {
  return 1.0f / (1.0f + __expf(-x));
}

// ---------------------------------------------------------------------------
// Kernel 1: pack weights/bias to bf16, gate-interleaved (p = 4*j + gate),
// K padded to 320 with zeros.
// ---------------------------------------------------------------------------
__global__ void pack_weights_kernel(const float* __restrict__ wih_f,
                                    const float* __restrict__ whh_f,
                                    const float* __restrict__ bih_f,
                                    const float* __restrict__ bhh_f,
                                    const float* __restrict__ wih_b,
                                    const float* __restrict__ whh_b,
                                    const float* __restrict__ bih_b,
                                    const float* __restrict__ bhh_b,
                                    __bf16* __restrict__ WihP,
                                    __bf16* __restrict__ WhhP,
                                    float*  __restrict__ BiasP) {
  const int tid = blockIdx.x * blockDim.x + threadIdx.x;
  const int total = 2 * G4 * DP;
  if (tid >= total) return;
  const int dir = tid / (G4 * DP);
  const int rem = tid % (G4 * DP);
  const int p   = rem / DP;             // packed row
  const int k   = rem % DP;
  const int j    = p >> 2;
  const int gate = p & 3;
  const int orow = gate * H_ + j;       // torch gate order i,f,g,o
  const float* wih = dir ? wih_b : wih_f;
  const float* whh = dir ? whh_b : whh_f;
  const float* bih = dir ? bih_b : bih_f;
  const float* bhh = dir ? bhh_b : bhh_f;
  const float wi = (k < D_) ? wih[orow * D_ + k] : 0.0f;
  const float wh = (k < H_) ? whh[orow * H_ + k] : 0.0f;
  WihP[tid] = (__bf16)wi;
  WhhP[tid] = (__bf16)wh;
  if (k == 0) BiasP[dir * G4 + p] = bih[orow] + bhh[orow];
}

// ---------------------------------------------------------------------------
// Kernel 2: embedding gather -> bf16, padded to DP.
// ---------------------------------------------------------------------------
__global__ void embed_kernel(const int* __restrict__ tokens,
                             const float* __restrict__ emb,
                             __bf16* __restrict__ Xbf) {
  const int tid = blockIdx.x * blockDim.x + threadIdx.x;
  if (tid >= MR * DP) return;
  const int row = tid / DP;      // b*T + t (physical time)
  const int d   = tid % DP;
  float v = 0.0f;
  if (d < D_) {
    const int tok = tokens[row];
    v = emb[(size_t)tok * D_ + d];
  }
  Xbf[tid] = (__bf16)v;
}

// ---------------------------------------------------------------------------
// Kernel 3: input projection GEMM, Gates[dir] = Xbf @ WihP[dir]^T.
// One wave per 16x16 output tile; K-chain of 10 x v_wmma_f32_16x16x32_bf16.
// A frag: 16-bit A 16x32 layout (lanes 0-15 K=0..7/16..23, 16-31 +8).
// B frag: same K layout as A (two 16B loads per lane from packed row-major W).
// ---------------------------------------------------------------------------
__global__ void input_gemm_kernel(const __bf16* __restrict__ Xbf,
                                  const __bf16* __restrict__ WihP,
                                  float* __restrict__ Gates) {
  const int lane = threadIdx.x;        // 32 threads / block = one wave
  const int mt   = blockIdx.x;         // 0..2047
  const int nt   = blockIdx.y;         // 0..74
  const int dir  = blockIdx.z;         // 0..1
  const int l15  = lane & 15;
  const int kh   = lane >> 4;          // K-half select

  const __bf16* Wd = WihP + (size_t)dir * G4 * DP;
  float* Gd = Gates + (size_t)dir * MR * G4;

  const size_t arow = (size_t)(mt * 16 + l15) * DP;
  const size_t brow = (size_t)(nt * 16 + l15) * DP;

  v8f acc = {0.f, 0.f, 0.f, 0.f, 0.f, 0.f, 0.f, 0.f};
#pragma unroll
  for (int kt = 0; kt < KT; ++kt) {
    const int k0 = kt * 32;
    // prefetch next K-slice of W into caches (global_prefetch_b8)
    if (kt + 1 < KT)
      __builtin_prefetch((const void*)(Wd + brow + k0 + 32), 0, 1);
    const v8bf alo = *(const v8bf*)(Xbf + arow + k0 + kh * 8);
    const v8bf ahi = *(const v8bf*)(Xbf + arow + k0 + 16 + kh * 8);
    const v16bf a = __builtin_shufflevector(alo, ahi,
        0, 1, 2, 3, 4, 5, 6, 7, 8, 9, 10, 11, 12, 13, 14, 15);
    const v8bf blo = *(const v8bf*)(Wd + brow + k0 + kh * 8);
    const v8bf bhi = *(const v8bf*)(Wd + brow + k0 + 16 + kh * 8);
    const v16bf b = __builtin_shufflevector(blo, bhi,
        0, 1, 2, 3, 4, 5, 6, 7, 8, 9, 10, 11, 12, 13, 14, 15);
    acc = __builtin_amdgcn_wmma_f32_16x16x32_bf16(
        false, a, false, b, (short)0, acc, false, false);
  }
  // C/D layout: VGPR r -> M = r (lanes 0-15) / 8+r (lanes 16-31), N = lane&15
  const int col = nt * 16 + l15;
#pragma unroll
  for (int r = 0; r < 8; ++r) {
    const int row = mt * 16 + r + kh * 8;
    Gd[(size_t)row * G4 + col] = acc[r];
  }
}

// ---------------------------------------------------------------------------
// Kernel 4: persistent recurrent LSTM. grid = (4 batch-tiles, 2 dirs),
// 256 threads = 8 waves. h (bf16) and c (f32) live in LDS; W_hh streamed
// from L2 every step. Wave w owns N-tiles {w, w+8, ...}; thanks to the
// gate-interleaved packing each tile holds 4 complete hidden units, so the
// owning wave finishes the LSTM cell locally via a 1KB LDS staging tile.
// ---------------------------------------------------------------------------
__global__ void lstm_recurrent_kernel(const __bf16* __restrict__ WhhP,
                                      const float* __restrict__ BiasP,
                                      const float* __restrict__ Gates,
                                      const float* __restrict__ h0,
                                      const float* __restrict__ c0,
                                      float* __restrict__ Hout) {
  __shared__ __align__(16) __bf16 sh_h[16][DP];
  __shared__ float sh_c[16][H_];
  __shared__ float stile[8][16][16];

  const int bt   = blockIdx.x;   // batch tile (16 rows)
  const int dir  = blockIdx.y;
  const int tidx = threadIdx.x;
  const int lane = tidx & 31;
  const int w    = tidx >> 5;    // wave id 0..7
  const int l15  = lane & 15;
  const int kh   = lane >> 4;

  const __bf16* Wd  = WhhP  + (size_t)dir * G4 * DP;
  const float*  bia = BiasP + (size_t)dir * G4;
  const float*  Gd  = Gates + (size_t)dir * MR * G4;
  float*        Hd  = Hout  + (size_t)dir * MR * H_;

  // init h, c from h0/c0
  for (int idx = tidx; idx < 16 * DP; idx += 256) {
    const int m = idx / DP, d = idx % DP;
    const int gb = bt * 16 + m;
    float hv = 0.0f;
    if (d < H_) {
      hv = h0[((size_t)dir * B_ + gb) * H_ + d];
      sh_c[m][d] = c0[((size_t)dir * B_ + gb) * H_ + d];
    }
    sh_h[m][d] = (__bf16)hv;
  }
  __syncthreads();

  const v8f vz = {0.f, 0.f, 0.f, 0.f, 0.f, 0.f, 0.f, 0.f};

#pragma unroll 1
  for (int t = 0; t < T_; ++t) {
    const int tphys = (dir == 0) ? t : (T_ - 1 - t);

    // ---- phase 1: acc[i] = h @ W_hh^T tile (reads sh_h) ----
    v8f acc[10];
#pragma unroll 1
    for (int i = 0; i < 10; ++i) {
      const int nt = w + i * 8;
      if (nt >= NT) break;
      v8f a = vz;
      const size_t brow = (size_t)(nt * 16 + l15) * DP;
#pragma unroll
      for (int kt = 0; kt < KT; ++kt) {
        const int k0 = kt * 32;
        const v8bf alo = *(const v8bf*)(&sh_h[l15][k0 + kh * 8]);
        const v8bf ahi = *(const v8bf*)(&sh_h[l15][k0 + 16 + kh * 8]);
        const v16bf af = __builtin_shufflevector(alo, ahi,
            0, 1, 2, 3, 4, 5, 6, 7, 8, 9, 10, 11, 12, 13, 14, 15);
        const v8bf blo = *(const v8bf*)(Wd + brow + k0 + kh * 8);
        const v8bf bhi = *(const v8bf*)(Wd + brow + k0 + 16 + kh * 8);
        const v16bf bf = __builtin_shufflevector(blo, bhi,
            0, 1, 2, 3, 4, 5, 6, 7, 8, 9, 10, 11, 12, 13, 14, 15);
        a = __builtin_amdgcn_wmma_f32_16x16x32_bf16(
            false, af, false, bf, (short)0, a, false, false);
      }
      acc[i] = a;
    }
    __syncthreads();   // all sh_h reads of step t complete

    // ---- phase 2: add gates_in + bias, LSTM cell, update sh_h/sh_c ----
#pragma unroll 1
    for (int i = 0; i < 10; ++i) {
      const int nt = w + i * 8;
      if (nt >= NT) break;
      const int col = nt * 16 + l15;
#pragma unroll
      for (int r = 0; r < 8; ++r) {
        const int m = r + kh * 8;
        const size_t grow = (size_t)((bt * 16 + m) * T_ + tphys) * G4;
        stile[w][m][l15] = acc[i][r] + Gd[grow + col] + bia[col];
      }
      __builtin_amdgcn_wave_barrier();
#pragma unroll
      for (int q = 0; q < 2; ++q) {
        const int p  = lane * 2 + q;   // 0..63 = 16 rows x 4 units
        const int m  = p >> 2;
        const int ju = p & 3;
        const float gi = stile[w][m][4 * ju + 0];
        const float gf = stile[w][m][4 * ju + 1];
        const float gg = stile[w][m][4 * ju + 2];
        const float go = stile[w][m][4 * ju + 3];
        const int j = nt * 4 + ju;     // hidden unit 0..299
        const float cn = sigf(gf) * sh_c[m][j] + sigf(gi) * tanhf(gg);
        const float hn = sigf(go) * tanhf(cn);
        sh_c[m][j] = cn;
        sh_h[m][j] = (__bf16)hn;
        Hd[(size_t)((bt * 16 + m) * T_ + tphys) * H_ + j] = hn;
      }
      __builtin_amdgcn_wave_barrier();
    }
    __syncthreads();   // new h visible before step t+1 reads
  }
}

// ---------------------------------------------------------------------------
// Kernel 5: emissions = [hf, hb] @ w_out^T + b_out.  N=9 -> plain VALU.
// ---------------------------------------------------------------------------
__global__ void emissions_kernel(const float* __restrict__ Hout,
                                 const float* __restrict__ w_out,
                                 const float* __restrict__ b_out,
                                 float* __restrict__ Emis) {
  const int tid = blockIdx.x * blockDim.x + threadIdx.x;
  if (tid >= MR * K_) return;
  const int row = tid / K_;      // b*T + t
  const int k   = tid % K_;
  const float* hf = Hout + (size_t)row * H_;
  const float* hb = Hout + (size_t)MR * H_ + (size_t)row * H_;
  const float* wk = w_out + (size_t)k * (2 * H_);
  float acc = b_out[k];
  for (int j = 0; j < H_; ++j)
    acc = fmaf(hf[j], wk[j], fmaf(hb[j], wk[H_ + j], acc));
  Emis[tid] = acc;
}

// ---------------------------------------------------------------------------
// Kernel 6: Viterbi (forward + backtrack). One thread per batch row (K=9).
// Paths written as float tag indices into d_out[0 .. B*T).
// ---------------------------------------------------------------------------
__global__ void viterbi_kernel(const float* __restrict__ Emis,
                               const unsigned char* __restrict__ mask,
                               const float* __restrict__ start_t,
                               const float* __restrict__ end_t,
                               const float* __restrict__ trans,
                               int* __restrict__ hist,
                               float* __restrict__ out) {
  const int b = blockIdx.x * blockDim.x + threadIdx.x;
  if (b >= B_) return;
  float score[K_];
  for (int k = 0; k < K_; ++k) score[k] = start_t[k] + Emis[(size_t)b * T_ * K_ + k];
  for (int t = 1; t < T_; ++t) {
    const float* et = Emis + ((size_t)b * T_ + t) * K_;
    float ns[K_];
    for (int kn = 0; kn < K_; ++kn) {
      float best = score[0] + trans[kn];
      int bi = 0;
      for (int kp = 1; kp < K_; ++kp) {
        const float v = score[kp] + trans[kp * K_ + kn];
        if (v > best) { best = v; bi = kp; }
      }
      hist[((size_t)(t - 1) * B_ + b) * K_ + kn] = bi;
      ns[kn] = best + et[kn];
    }
    if (mask[b * T_ + t])
      for (int k = 0; k < K_; ++k) score[k] = ns[k];
  }
  int last = 0;
  float best = score[0] + end_t[0];
  for (int k = 1; k < K_; ++k) {
    const float v = score[k] + end_t[k];
    if (v > best) { best = v; last = k; }
  }
  int tag = last;
  out[b * T_ + (T_ - 1)] = (float)tag;
  for (int t = T_ - 2; t >= 0; --t) {
    const int prev = hist[((size_t)t * B_ + b) * K_ + tag];
    if (mask[b * T_ + t + 1]) tag = prev;
    out[b * T_ + t] = (float)tag;
  }
}

// ---------------------------------------------------------------------------
// Kernel 7: CRF NLL (gold-path score + forward logsumexp), atomic reduce.
// ---------------------------------------------------------------------------
__global__ void zero_loss_kernel(float* loss) { *loss = 0.0f; }

__global__ void crf_nll_kernel(const float* __restrict__ Emis,
                               const int* __restrict__ tags,
                               const unsigned char* __restrict__ mask,
                               const float* __restrict__ start_t,
                               const float* __restrict__ end_t,
                               const float* __restrict__ trans,
                               float* __restrict__ loss) {
  const int b = blockIdx.x * blockDim.x + threadIdx.x;
  if (b >= B_) return;
  const float* eb = Emis + (size_t)b * T_ * K_;
  // numerator
  int prev = tags[b * T_];
  float num = start_t[prev] + eb[prev];
  int nmask = mask[b * T_] ? 1 : 0;
  for (int t = 1; t < T_; ++t) {
    const int tt = tags[b * T_ + t];
    const float mt = mask[b * T_ + t] ? 1.0f : 0.0f;
    nmask += mask[b * T_ + t] ? 1 : 0;
    num += (trans[prev * K_ + tt] + eb[t * K_ + tt]) * mt;
    prev = tt;
  }
  const int last_idx = nmask - 1;
  num += end_t[tags[b * T_ + last_idx]];
  // denominator: forward algorithm
  float alpha[K_];
  for (int k = 0; k < K_; ++k) alpha[k] = start_t[k] + eb[k];
  for (int t = 1; t < T_; ++t) {
    float na[K_];
    for (int kn = 0; kn < K_; ++kn) {
      float mx = -1e30f;
      for (int kp = 0; kp < K_; ++kp)
        mx = fmaxf(mx, alpha[kp] + trans[kp * K_ + kn]);
      float s = 0.0f;
      for (int kp = 0; kp < K_; ++kp)
        s += __expf(alpha[kp] + trans[kp * K_ + kn] - mx);
      na[kn] = mx + __logf(s) + eb[t * K_ + kn];
    }
    if (mask[b * T_ + t])
      for (int k = 0; k < K_; ++k) alpha[k] = na[k];
  }
  float mx = -1e30f;
  for (int k = 0; k < K_; ++k) mx = fmaxf(mx, alpha[k] + end_t[k]);
  float s = 0.0f;
  for (int k = 0; k < K_; ++k) s += __expf(alpha[k] + end_t[k] - mx);
  const float den = mx + __logf(s);
  atomicAdd(loss, -(num - den));
}

// ---------------------------------------------------------------------------
static inline size_t align256(size_t x) { return (x + 255) & ~(size_t)255; }

extern "C" void kernel_launch(void* const* d_in, const int* in_sizes, int n_in,
                              void* d_out, int out_size, void* d_ws, size_t ws_size,
                              hipStream_t stream) {
  (void)in_sizes; (void)n_in; (void)out_size; (void)ws_size;
  const int*   tokens  = (const int*)  d_in[0];
  const int*   tags    = (const int*)  d_in[1];
  const unsigned char* mask = (const unsigned char*)d_in[2];
  const float* emb     = (const float*)d_in[3];
  const float* w_ih_f  = (const float*)d_in[4];
  const float* w_hh_f  = (const float*)d_in[5];
  const float* b_ih_f  = (const float*)d_in[6];
  const float* b_hh_f  = (const float*)d_in[7];
  const float* w_ih_b  = (const float*)d_in[8];
  const float* w_hh_b  = (const float*)d_in[9];
  const float* b_ih_b  = (const float*)d_in[10];
  const float* b_hh_b  = (const float*)d_in[11];
  const float* h0      = (const float*)d_in[12];
  const float* c0      = (const float*)d_in[13];
  const float* w_out   = (const float*)d_in[14];
  const float* b_out   = (const float*)d_in[15];
  const float* start_t = (const float*)d_in[16];
  const float* end_t   = (const float*)d_in[17];
  const float* trans   = (const float*)d_in[18];

  char* ws = (char*)d_ws;
  size_t off = 0;
  __bf16* WihP  = (__bf16*)(ws + off); off = align256(off + (size_t)2 * G4 * DP * 2);
  __bf16* WhhP  = (__bf16*)(ws + off); off = align256(off + (size_t)2 * G4 * DP * 2);
  float*  BiasP = (float*) (ws + off); off = align256(off + (size_t)2 * G4 * 4);
  __bf16* Xbf   = (__bf16*)(ws + off); off = align256(off + (size_t)MR * DP * 2);
  float*  Gates = (float*) (ws + off); off = align256(off + (size_t)2 * MR * G4 * 4);
  float*  Hout  = (float*) (ws + off); off = align256(off + (size_t)2 * MR * H_ * 4);
  float*  Emis  = (float*) (ws + off); off = align256(off + (size_t)MR * K_ * 4);
  int*    Hist  = (int*)   (ws + off); off = align256(off + (size_t)(T_ - 1) * B_ * K_ * 4);

  float* paths = (float*)d_out;            // B*T path entries
  float* loss  = (float*)d_out + MR;       // scalar loss

  // 1. pack weights
  {
    const int total = 2 * G4 * DP;
    pack_weights_kernel<<<(total + 255) / 256, 256, 0, stream>>>(
        w_ih_f, w_hh_f, b_ih_f, b_hh_f, w_ih_b, w_hh_b, b_ih_b, b_hh_b,
        WihP, WhhP, BiasP);
  }
  // 2. embedding gather -> bf16
  {
    const int total = MR * DP;
    embed_kernel<<<(total + 255) / 256, 256, 0, stream>>>(tokens, emb, Xbf);
  }
  // 3. input projection GEMMs (both directions), WMMA bf16
  {
    dim3 grid(MR / 16, NT, 2);
    input_gemm_kernel<<<grid, 32, 0, stream>>>(Xbf, WihP, Gates);
  }
  // 4. persistent recurrent LSTM (4 batch tiles x 2 directions)
  {
    dim3 grid(B_ / 16, 2);
    lstm_recurrent_kernel<<<grid, 256, 0, stream>>>(WhhP, BiasP, Gates, h0, c0, Hout);
  }
  // 5. emissions
  {
    const int total = MR * K_;
    emissions_kernel<<<(total + 255) / 256, 256, 0, stream>>>(Hout, w_out, b_out, Emis);
  }
  // 6. viterbi
  viterbi_kernel<<<1, B_, 0, stream>>>(Emis, mask, start_t, end_t, trans, Hist, paths);
  // 7. CRF NLL
  zero_loss_kernel<<<1, 1, 0, stream>>>(loss);
  crf_nll_kernel<<<1, B_, 0, stream>>>(Emis, tags, mask, start_t, end_t, trans, loss);
}
